// EncoderLayer_5282809774904
// MI455X (gfx1250) — matrix operand order repaired and checked by reading, not compiled, not run on hardware
//
#include <hip/hip_runtime.h>
#include <hip/hip_bf16.h>

// ---------------------------------------------------------------------------
// Encoder layer for MI455X (gfx1250, wave32, WMMA).
//   x2 = instnorm(x)
//   x  = x + MHA_sa(x2) + MHA_ga(x2)
//   x2 = instnorm(x)
//   out = x + relu(x2@W1+b1)@W2 + b2
// All GEMMs: v_wmma_f32_16x16x32_bf16 (fp32 accum). Weights/activations
// repacked to bf16 fragment-native layouts so every WMMA operand is loaded
// with 128-bit vector loads.
// ---------------------------------------------------------------------------

#define SEQ     2048
#define BS      2
#define TOKENS  (BS*SEQ)      // 4096
#define DM      256
#define DFF     512
#define NH      8
#define DK      32
#define QSCALE  0.17677669529663687f   // 1/sqrt(32)

typedef __attribute__((ext_vector_type(16))) __bf16 v16bf;
typedef __attribute__((ext_vector_type(8)))  float  v8f;
typedef unsigned short u16;
typedef unsigned int   u32;

union FragA { v16bf v; uint4 q[2]; u16 u[16]; };
union FragC { v8f  v; float f[8]; };

__device__ inline u16 f2bf(float f) {            // round-to-nearest-even
  union { float f; u32 u; } c; c.f = f;
  u32 u = c.u + 0x7fffu + ((c.u >> 16) & 1u);
  return (u16)(u >> 16);
}

__device__ inline v8f zerov8() {
  FragC z;
#pragma unroll
  for (int i = 0; i < 8; ++i) z.f[i] = 0.f;
  return z.v;
}

__device__ inline v8f wmma_bf16(v16bf a, v16bf b, v8f c) {
  return __builtin_amdgcn_wmma_f32_16x16x32_bf16(false, a, false, b,
                                                 (short)0, c, false, false);
}

// A fragment (16x32 bf16, row-major, ld elements per row).
// lane L: row = L%16; elems 0..15 = K[(L/16)*16 .. +15].
__device__ inline v16bf load_a_frag(const u16* __restrict__ p, int ld) {
  int lane = threadIdx.x & 31;
  const u16* row = p + (size_t)(lane & 15) * ld + (lane >> 4) * 16;
  FragA f;
  f.q[0] = *(const uint4*)(row);
  f.q[1] = *(const uint4*)(row + 8);
  return f.v;
}

// B fragment from fragment-native packed tile (512 bf16 per 32x16 tile):
// pack[lane*16 + i] = B[(lane/16)*16 + i][lane%16]
__device__ inline v16bf load_b_packed(const u16* __restrict__ p) {
  int lane = threadIdx.x & 31;
  FragA f;
  const uint4* s = (const uint4*)(p + lane * 16);
  f.q[0] = s[0]; f.q[1] = s[1];
  return f.v;
}

// ---------------------------------------------------------------------------
// Repack fp32 weight (Ktot x Ntot, row-major) -> bf16 packed B tiles.
// tile index = (no*KT + ko); within tile: lane*16+i as above.
__global__ void k_repack(const float* __restrict__ w, u16* __restrict__ dst,
                         int KT, int Ntot) {
  int idx  = blockIdx.x * 256 + threadIdx.x;
  int tile = idx >> 9, r = idx & 511;
  int lane = r >> 4, i = r & 15;
  int ko = tile % KT, no = tile / KT;
  int k  = ko * 32 + (lane >> 4) * 16 + i;
  int n  = no * 16 + (lane & 15);
  dst[idx] = f2bf(w[(size_t)k * Ntot + n]);
}

// ---------------------------------------------------------------------------
// Instance norm over last dim (256) -> bf16. One wave per token.
__global__ void k_norm(const float* __restrict__ x, u16* __restrict__ o) {
  int wave = threadIdx.x >> 5, lane = threadIdx.x & 31;
  int tok  = blockIdx.x * 8 + wave;
  const float4* p = (const float4*)(x + (size_t)tok * DM) + lane * 2;
  float4 a = p[0], b = p[1];
  float s = a.x + a.y + a.z + a.w + b.x + b.y + b.z + b.w;
#pragma unroll
  for (int m = 1; m < 32; m <<= 1) s += __shfl_xor(s, m, 32);
  float mu = s * (1.f / 256.f);
  float e[8] = { a.x - mu, a.y - mu, a.z - mu, a.w - mu,
                 b.x - mu, b.y - mu, b.z - mu, b.w - mu };
  float v = 0.f;
#pragma unroll
  for (int i = 0; i < 8; ++i) v += e[i] * e[i];
#pragma unroll
  for (int m = 1; m < 32; m <<= 1) v += __shfl_xor(v, m, 32);
  float rs = rsqrtf(v * (1.f / 256.f) + 1e-5f);
  union { uint4 q; u16 u[8]; } ob;
#pragma unroll
  for (int i = 0; i < 8; ++i) ob.u[i] = f2bf(e[i] * rs);
  *(uint4*)(o + (size_t)tok * DM + lane * 8) = ob.q;
}

// ---------------------------------------------------------------------------
// QKV projections for both attentions. blockIdx.y in 0..5 selects
// {sa_q, sa_k, sa_v, ga_q, ga_k, ga_v}. Q: scaled by 1/sqrt(dk), layout
// (b,h,s,dk) row-major (A-fragments for scores). K,V: fragment-native packed.
struct ProjArgs {
  const u16* xn;
  const u16* wp[6];
  const float* bias[6];
  u16* dst[6];
};

__global__ void k_proj(ProjArgs A) {
  int wave = threadIdx.x >> 5, lane = threadIdx.x & 31;
  int z    = blockIdx.y;
  int id   = blockIdx.x * 8 + wave;        // 0..4095
  int mtile = id >> 4, ntile = id & 15;
  const u16*  wp   = A.wp[z];
  const float* bias = A.bias[z];
  u16*  dst  = A.dst[z];
  int mode   = z % 3;                      // 0=Q 1=K 2=V

  v8f c = zerov8();
#pragma unroll
  for (int ko = 0; ko < 8; ++ko) {
    v16bf a = load_a_frag(A.xn + (size_t)mtile * 16 * DM + ko * 32, DM);
    v16bf b = load_b_packed(wp + (size_t)(ntile * 8 + ko) * 512);
    c = wmma_bf16(a, b, c);
  }
  int n = lane & 15, hi = lane >> 4;
  int ng = ntile * 16 + n;
  float bv = bias[ng];
  int h = ng >> 5, d = ng & 31;
  FragC fc; fc.v = c;
#pragma unroll
  for (int i = 0; i < 8; ++i) {
    int tok = mtile * 16 + i + 8 * hi;
    int b_  = tok >> 11, t = tok & (SEQ - 1);
    int bh  = b_ * NH + h;
    float val = fc.f[i] + bv;
    size_t idx;
    if (mode == 0) { val *= QSCALE; idx = ((size_t)bh * SEQ + t) * DK + d; }
    else if (mode == 1)
      idx = (size_t)bh * 65536 + (size_t)(t >> 4) * 512
          + ((d >> 4) * 16 + (t & 15)) * 16 + (d & 15);
    else
      idx = (size_t)bh * 65536 + ((size_t)(d >> 4) * 64 + (t >> 5)) * 512
          + ((((t >> 4) & 1) * 16 + (d & 15)) * 16) + (t & 15);
    dst[idx] = f2bf(val);
  }
}

// ---------------------------------------------------------------------------
// Flash attention: one wave owns a 16-query tile; online softmax across 2048
// keys in 32-key steps. Score tile = 1 WMMA (K=32=d_k). P goes through LDS
// (layout swap D->A). Output (bf16) written into (token, 256) row-major.
__global__ void k_attn(const u16* __restrict__ q, const u16* __restrict__ kp,
                       const u16* __restrict__ vp, const int* __restrict__ mask,
                       u16* __restrict__ ao) {
  __shared__ u16 pl[8][512];
  int wave = threadIdx.x >> 5, lane = threadIdx.x & 31;
  int id = blockIdx.x * 8 + wave;          // 0..2047
  int bh = id >> 7, qt = id & 127;
  int b_ = bh >> 3, h = bh & 7;
  int qbase = qt * 16;
  int n = lane & 15, hi = lane >> 4;

  v16bf aq = load_a_frag(q + ((size_t)bh * SEQ + qbase) * DK, DK);
  const int* mbase = mask + (size_t)b_ * SEQ * SEQ;
  const u16* kbh = kp + (size_t)bh * 65536;
  const u16* vbh = vp + (size_t)bh * 65536;

  v8f acc0 = zerov8(), acc1 = zerov8();
  float m8[8], l8[8];
#pragma unroll
  for (int i = 0; i < 8; ++i) { m8[i] = -3.0e38f; l8[i] = 0.f; }

  for (int kb = 0; kb < 64; ++kb) {
    int k0 = kb * 32;
    v8f zc = zerov8();
    FragC s0, s1;
    s0.v = wmma_bf16(aq, load_b_packed(kbh + (size_t)(k0 >> 4) * 512), zc);
    s1.v = wmma_bf16(aq, load_b_packed(kbh + (size_t)((k0 >> 4) + 1) * 512), zc);

#pragma unroll
    for (int i = 0; i < 8; ++i) {
      int qr = qbase + i + 8 * hi;
      const int* mr = mbase + (size_t)qr * SEQ + k0 + n;
      int mv0 = mr[0], mv1 = mr[16];
      if (kb < 63) __builtin_prefetch(mr + 32, 0, 1);   // global_prefetch_b8
      float v0 = (mv0 == 1) ? -1.0e9f : s0.f[i];
      float v1 = (mv1 == 1) ? -1.0e9f : s1.f[i];
      float rm = fmaxf(v0, v1);
      rm = fmaxf(rm, __shfl_xor(rm, 1, 32));
      rm = fmaxf(rm, __shfl_xor(rm, 2, 32));
      rm = fmaxf(rm, __shfl_xor(rm, 4, 32));
      rm = fmaxf(rm, __shfl_xor(rm, 8, 32));
      float mn = fmaxf(m8[i], rm);
      float sc = __expf(m8[i] - mn);
      float p0 = __expf(v0 - mn);
      float p1 = __expf(v1 - mn);
      float rs = p0 + p1;
      rs += __shfl_xor(rs, 1, 32);
      rs += __shfl_xor(rs, 2, 32);
      rs += __shfl_xor(rs, 4, 32);
      rs += __shfl_xor(rs, 8, 32);
      l8[i] = l8[i] * sc + rs;
      m8[i] = mn;
      acc0[i] *= sc; acc1[i] *= sc;
      int r = i + 8 * hi;                  // D-layout row -> LDS row-major
      pl[wave][r * 32 + n]      = f2bf(p0);
      pl[wave][r * 32 + 16 + n] = f2bf(p1);
    }
    asm volatile("s_wait_dscnt 0" ::: "memory");
    FragA fp;
    {
      const u16* row = &pl[wave][(lane & 15) * 32 + hi * 16];
      fp.q[0] = *(const uint4*)(row);
      fp.q[1] = *(const uint4*)(row + 8);
    }
    acc0 = wmma_bf16(fp.v, load_b_packed(vbh + (size_t)kb * 512), acc0);
    acc1 = wmma_bf16(fp.v, load_b_packed(vbh + (size_t)(64 + kb) * 512), acc1);
  }

#pragma unroll
  for (int i = 0; i < 8; ++i) {
    float inv = 1.0f / l8[i];
    int tok = (b_ << 11) + qbase + i + 8 * hi;
    u16* o = ao + (size_t)tok * DM + h * DK;
    o[n]      = f2bf(acc0[i] * inv);
    o[16 + n] = f2bf(acc1[i] * inv);
  }
}

// ---------------------------------------------------------------------------
// y = x + Osa@Wo_sa + bo_sa + Oga@Wo_ga + bo_ga   (fp32 result kept for
// residual + second norm)
__global__ void k_combine(const u16* __restrict__ aoS, const u16* __restrict__ aoG,
                          const u16* __restrict__ woS, const u16* __restrict__ woG,
                          const float* __restrict__ boS, const float* __restrict__ boG,
                          const float* __restrict__ x, float* __restrict__ y) {
  int wave = threadIdx.x >> 5, lane = threadIdx.x & 31;
  int id = blockIdx.x * 8 + wave;
  int mtile = id >> 4, ntile = id & 15;
  v8f c = zerov8();
#pragma unroll
  for (int ko = 0; ko < 8; ++ko) {
    c = wmma_bf16(load_a_frag(aoS + (size_t)mtile * 16 * DM + ko * 32, DM),
                  load_b_packed(woS + (size_t)(ntile * 8 + ko) * 512), c);
    c = wmma_bf16(load_a_frag(aoG + (size_t)mtile * 16 * DM + ko * 32, DM),
                  load_b_packed(woG + (size_t)(ntile * 8 + ko) * 512), c);
  }
  int n = lane & 15, hi = lane >> 4;
  int ng = ntile * 16 + n;
  float bb = boS[ng] + boG[ng];
  FragC fc; fc.v = c;
#pragma unroll
  for (int i = 0; i < 8; ++i) {
    size_t tok = mtile * 16 + i + 8 * hi;
    y[tok * DM + ng] = x[tok * DM + ng] + fc.f[i] + bb;
  }
}

// h1 = relu(yn @ W1 + b1), bf16
__global__ void k_ffn1(const u16* __restrict__ yn, const u16* __restrict__ w1p,
                       const float* __restrict__ b1, u16* __restrict__ h1) {
  int wave = threadIdx.x >> 5, lane = threadIdx.x & 31;
  int id = blockIdx.x * 8 + wave;          // 0..8191
  int mtile = id >> 5, ntile = id & 31;
  v8f c = zerov8();
#pragma unroll
  for (int ko = 0; ko < 8; ++ko)
    c = wmma_bf16(load_a_frag(yn + (size_t)mtile * 16 * DM + ko * 32, DM),
                  load_b_packed(w1p + (size_t)(ntile * 8 + ko) * 512), c);
  int n = lane & 15, hi = lane >> 4;
  int ng = ntile * 16 + n;
  float bb = b1[ng];
  FragC fc; fc.v = c;
#pragma unroll
  for (int i = 0; i < 8; ++i) {
    size_t tok = mtile * 16 + i + 8 * hi;
    h1[tok * DFF + ng] = f2bf(fmaxf(fc.f[i] + bb, 0.f));
  }
}

// out = y + h1 @ W2 + b2  (fp32)
__global__ void k_ffn2(const u16* __restrict__ h1, const u16* __restrict__ w2p,
                       const float* __restrict__ b2, const float* __restrict__ y,
                       float* __restrict__ out) {
  int wave = threadIdx.x >> 5, lane = threadIdx.x & 31;
  int id = blockIdx.x * 8 + wave;
  int mtile = id >> 4, ntile = id & 15;
  v8f c = zerov8();
#pragma unroll
  for (int ko = 0; ko < 16; ++ko)
    c = wmma_bf16(load_a_frag(h1 + (size_t)mtile * 16 * DFF + ko * 32, DFF),
                  load_b_packed(w2p + (size_t)(ntile * 16 + ko) * 512), c);
  int n = lane & 15, hi = lane >> 4;
  int ng = ntile * 16 + n;
  float bb = b2[ng];
  FragC fc; fc.v = c;
#pragma unroll
  for (int i = 0; i < 8; ++i) {
    size_t tok = mtile * 16 + i + 8 * hi;
    out[tok * DM + ng] = y[tok * DM + ng] + fc.f[i] + bb;
  }
}

// ---------------------------------------------------------------------------
extern "C" void kernel_launch(void* const* d_in, const int* in_sizes, int n_in,
                              void* d_out, int out_size, void* d_ws, size_t ws_size,
                              hipStream_t stream) {
  const float* x     = (const float*)d_in[0];
  const int*   smask = (const int*)d_in[1];
  const int*   gmask = (const int*)d_in[2];
  const float* sa_wq = (const float*)d_in[3];  const float* sa_bq = (const float*)d_in[4];
  const float* sa_wk = (const float*)d_in[5];  const float* sa_bk = (const float*)d_in[6];
  const float* sa_wv = (const float*)d_in[7];  const float* sa_bv = (const float*)d_in[8];
  const float* sa_wo = (const float*)d_in[9];  const float* sa_bo = (const float*)d_in[10];
  const float* ga_wq = (const float*)d_in[11]; const float* ga_bq = (const float*)d_in[12];
  const float* ga_wk = (const float*)d_in[13]; const float* ga_bk = (const float*)d_in[14];
  const float* ga_wv = (const float*)d_in[15]; const float* ga_bv = (const float*)d_in[16];
  const float* ga_wo = (const float*)d_in[17]; const float* ga_bo = (const float*)d_in[18];
  const float* w1    = (const float*)d_in[19]; const float* b1    = (const float*)d_in[20];
  const float* w2    = (const float*)d_in[21]; const float* b2    = (const float*)d_in[22];

  char* w = (char*)d_ws;
  auto alloc = [&](size_t bytes) -> char* {
    char* p = w; w += (bytes + 255) & ~(size_t)255; return p;
  };
  u16* xn  = (u16*)alloc((size_t)TOKENS * DM * 2);
  u16* qS  = (u16*)alloc((size_t)TOKENS * DM * 2);
  u16* qG  = (u16*)alloc((size_t)TOKENS * DM * 2);
  u16* kpS = (u16*)alloc((size_t)TOKENS * DM * 2);
  u16* kpG = (u16*)alloc((size_t)TOKENS * DM * 2);
  u16* vpS = (u16*)alloc((size_t)TOKENS * DM * 2);
  u16* vpG = (u16*)alloc((size_t)TOKENS * DM * 2);
  u16* aoS = (u16*)alloc((size_t)TOKENS * DM * 2);
  u16* aoG = (u16*)alloc((size_t)TOKENS * DM * 2);
  u16* yn  = (u16*)alloc((size_t)TOKENS * DM * 2);
  u16* h1  = (u16*)alloc((size_t)TOKENS * DFF * 2);
  float* y = (float*)alloc((size_t)TOKENS * DM * 4);
  u16* wpSq = (u16*)alloc(65536 * 2); u16* wpSk = (u16*)alloc(65536 * 2);
  u16* wpSv = (u16*)alloc(65536 * 2); u16* wpSo = (u16*)alloc(65536 * 2);
  u16* wpGq = (u16*)alloc(65536 * 2); u16* wpGk = (u16*)alloc(65536 * 2);
  u16* wpGv = (u16*)alloc(65536 * 2); u16* wpGo = (u16*)alloc(65536 * 2);
  u16* w1p  = (u16*)alloc(131072 * 2); u16* w2p = (u16*)alloc(131072 * 2);

  // Weight repack (bf16, fragment-native B layout)
  k_repack<<<256, 256, 0, stream>>>(sa_wq, wpSq, 8, 256);
  k_repack<<<256, 256, 0, stream>>>(sa_wk, wpSk, 8, 256);
  k_repack<<<256, 256, 0, stream>>>(sa_wv, wpSv, 8, 256);
  k_repack<<<256, 256, 0, stream>>>(sa_wo, wpSo, 8, 256);
  k_repack<<<256, 256, 0, stream>>>(ga_wq, wpGq, 8, 256);
  k_repack<<<256, 256, 0, stream>>>(ga_wk, wpGk, 8, 256);
  k_repack<<<256, 256, 0, stream>>>(ga_wv, wpGv, 8, 256);
  k_repack<<<256, 256, 0, stream>>>(ga_wo, wpGo, 8, 256);
  k_repack<<<512, 256, 0, stream>>>(w1, w1p, 8, 512);
  k_repack<<<512, 256, 0, stream>>>(w2, w2p, 16, 256);

  // Norm 1
  k_norm<<<512, 256, 0, stream>>>(x, xn);

  // QKV projections (both attentions)
  ProjArgs pa;
  pa.xn = xn;
  pa.wp[0] = wpSq; pa.wp[1] = wpSk; pa.wp[2] = wpSv;
  pa.wp[3] = wpGq; pa.wp[4] = wpGk; pa.wp[5] = wpGv;
  pa.bias[0] = sa_bq; pa.bias[1] = sa_bk; pa.bias[2] = sa_bv;
  pa.bias[3] = ga_bq; pa.bias[4] = ga_bk; pa.bias[5] = ga_bv;
  pa.dst[0] = qS; pa.dst[1] = kpS; pa.dst[2] = vpS;
  pa.dst[3] = qG; pa.dst[4] = kpG; pa.dst[5] = vpG;
  k_proj<<<dim3(512, 6), 256, 0, stream>>>(pa);

  // Flash attention (self + gen)
  k_attn<<<256, 256, 0, stream>>>(qS, kpS, vpS, smask, aoS);
  k_attn<<<256, 256, 0, stream>>>(qG, kpG, vpG, gmask, aoG);

  // Output projections + residual
  k_combine<<<512, 256, 0, stream>>>(aoS, aoG, wpSo, wpGo, sa_bo, ga_bo, x, y);

  // Norm 2 + FFN + residual
  k_norm<<<512, 256, 0, stream>>>(y, yn);
  k_ffn1<<<1024, 256, 0, stream>>>(yn, w1p, b1, h1);
  k_ffn2<<<512, 256, 0, stream>>>(h1, w2p, b2, y, (float*)d_out);
}
